// ALW_55611236548963
// MI455X (gfx1250) — matrix operand order repaired and checked by reading, not compiled, not run on hardware
//
#include <hip/hip_runtime.h>
#include <math.h>

typedef _Float16 v16h __attribute__((ext_vector_type(16)));
typedef _Float16 v8h  __attribute__((ext_vector_type(8)));
typedef float    v8f  __attribute__((ext_vector_type(8)));

#define B_    128
#define SEQ_  720
#define CH_   321
#define BC_   (B_*CH_)      // 41088
#define CPAD_ 384           // CH padded to multiple of 64 (zero-filled beyond 321)
#define LLOUT 104           // final ll length

// db6 analysis filters (H0 = db6, H1 = QMF)
__device__ const float H0c[12] = {
  0.11154074335008017f, 0.4946238903983854f, 0.7511339080215775f,
  0.3152503517092432f, -0.22626469396516913f, -0.12976686756709563f,
  0.09750160558707936f, 0.02752286553001629f, -0.031582039318031156f,
  0.0005538422009938016f, 0.004777257511010651f, -0.00107730108499558f };
__device__ const float H1c[12] = {
 -0.00107730108499558f, -0.004777257511010651f, 0.0005538422009938016f,
  0.031582039318031156f, 0.02752286553001629f, -0.09750160558707936f,
 -0.12976686756709563f, 0.22626469396516913f, 0.3152503517092432f,
 -0.7511339080215775f, 0.4946238903983854f, -0.11154074335008017f };

// ---------------- normalization + transpose ----------------
__global__ void norm_kernel(const float* __restrict__ x, float* __restrict__ xc,
                            float* __restrict__ meano, float* __restrict__ stdo) {
  int idx = blockIdx.x * blockDim.x + threadIdx.x;
  if (idx >= BC_) return;
  int b = idx / CH_, ch = idx % CH_;
  const float* base = x + (size_t)b * SEQ_ * CH_ + ch;
  float s = 0.f, s2 = 0.f;
  for (int n = 0; n < SEQ_; ++n) { float v = base[(size_t)n * CH_]; s += v; s2 += v * v; }
  float mean = s / SEQ_;
  float var = s2 / SEQ_ - mean * mean;
  float sd = sqrtf(var + 1e-5f);
  meano[idx] = mean; stdo[idx] = sd;
  float inv = 1.f / sd;
  float* orow = xc + (size_t)idx * SEQ_;
  for (int n = 0; n < SEQ_; ++n) orow[n] = (base[(size_t)n * CH_] - mean) * inv;
}

// ---------------- one DWT analysis level (symmetric pad, stride 2) ----------------
__global__ void dwt_kernel(const float* __restrict__ in, int N, int outN, int left,
                           float* __restrict__ lo, float* __restrict__ hi) {
  size_t idx = (size_t)blockIdx.x * blockDim.x + threadIdx.x;
  if (idx >= (size_t)BC_ * outN) return;
  int o = (int)(idx % outN); size_t bc = idx / outN;
  const float* row = in + bc * N;
  float a0 = 0.f, a1 = 0.f;
  #pragma unroll
  for (int t = 0; t < 12; ++t) {
    int j = 2 * o + t - left;
    int jj = (j < 0) ? (-j - 1) : ((j >= N) ? (2 * N - 1 - j) : j);
    float v = row[jj];
    a0 += v * H0c[t]; a1 += v * H1c[t];
  }
  lo[bc * outN + o] = a0; hi[bc * outN + o] = a1;
}

// ---------------- pack comp f32 [BC,L] -> f16 [BC,Lpad] (zero pad) ----------------
__global__ void pack_a_kernel(const float* __restrict__ src, int L, int Lpad,
                              _Float16* __restrict__ dst) {
  size_t idx = (size_t)blockIdx.x * blockDim.x + threadIdx.x;
  if (idx >= (size_t)BC_ * Lpad) return;
  int col = (int)(idx % Lpad); size_t row = idx / Lpad;
  dst[idx] = (col < L) ? (_Float16)src[row * L + col] : (_Float16)0.f;
}

// ---- pack weight w[L,L] (out = comp @ w^T) into B-fragment layout ----
// B[k,n] = w[n*L+k]; layout [kt][nt][lane][e], K = (e&7) + 8*(lane>>4) + ((e&8)?16:0), n = nt*16 + (lane&15)
__global__ void pack_w_kernel(const float* __restrict__ w, int L, int Lpad,
                              _Float16* __restrict__ wp) {
  int total = Lpad * Lpad;
  int idx = blockIdx.x * blockDim.x + threadIdx.x;
  if (idx >= total) return;
  int Ntiles = Lpad >> 4;
  int e = idx & 15, lane = (idx >> 4) & 31, tile = idx >> 9;
  int nt = tile % Ntiles, kt = tile / Ntiles;
  int h = lane >> 4;
  int n = nt * 16 + (lane & 15);
  int k = kt * 32 + (e & 7) + 8 * h + ((e & 8) ? 16 : 0);
  wp[idx] = (k < L && n < L) ? (_Float16)w[(size_t)n * L + k] : (_Float16)0.f;
}

// one K-step of the Q/K-linear GEMM: load A frag (direct) + B frag (pre-packed), WMMA
__device__ __forceinline__ v8f gemm_step(const _Float16* __restrict__ Ah, size_t rowbase,
                                         const _Float16* __restrict__ Wp, int Ntiles,
                                         int nt, int k0, int h, int lane, v8f acc) {
  v8h alo = *(const v8h*)(Ah + rowbase + k0 + 8 * h);        // e=0..7  : K=k0+8h+e
  v8h ahi = *(const v8h*)(Ah + rowbase + k0 + 16 + 8 * h);   // e=8..15 : K=k0+16+8h+(e-8)
  v16h a;
  #pragma unroll
  for (int i = 0; i < 8; ++i) { a[i] = alo[i]; a[i + 8] = ahi[i]; }
  v16h bf = *(const v16h*)(Wp + ((size_t)((k0 >> 5) * Ntiles + nt)) * 512 + lane * 16);
  return __builtin_amdgcn_wmma_f32_16x16x32_f16(false, a, false, bf, (short)0, acc, false, false);
}

// ---------------- Q/K linear: Out = tanh(A @ B + bias), f16 out, WMMA ----------------
// Dual accumulators (even/odd K tiles) to break the WMMA->WMMA RAW chain (kills hazard nops).
__global__ void gemm_tanh_kernel(const _Float16* __restrict__ Ah,
                                 const _Float16* __restrict__ Wp,
                                 const float* __restrict__ bias,
                                 int L, int Lpad, _Float16* __restrict__ Out) {
  const int w = threadIdx.x >> 5, lane = threadIdx.x & 31;
  const int h = lane >> 4, mn = lane & 15;
  const int m0 = (blockIdx.x * 8 + w) * 16;
  const int n0 = blockIdx.y * 16;
  const int Ntiles = Lpad >> 4;
  v8f acc0 = {}, acc1 = {};
  const size_t rowbase = (size_t)(m0 + mn) * Lpad;
  for (int k0 = 0; k0 < Lpad; k0 += 64) {   // Lpad is always a multiple of 64
    acc0 = gemm_step(Ah, rowbase, Wp, Ntiles, blockIdx.y, k0,      h, lane, acc0);
    acc1 = gemm_step(Ah, rowbase, Wp, Ntiles, blockIdx.y, k0 + 32, h, lane, acc1);
  }
  const int n = n0 + mn;
  const float bv = (n < L) ? bias[n] : 0.f;
  #pragma unroll
  for (int r = 0; r < 8; ++r) {
    int m = m0 + r + 8 * h;
    float v = (n < L) ? tanhf(acc0[r] + acc1[r] + bv) : 0.f;
    Out[(size_t)m * Lpad + n] = (_Float16)v;
  }
}

// one c-step of the scores GEMM: stage LDS tiles, gather frags, WMMA
__device__ __forceinline__ v8f scores_step(const _Float16* __restrict__ Qh,
                                           const _Float16* __restrict__ Kh,
                                           _Float16* __restrict__ qt, _Float16* __restrict__ kt,
                                           int b, int c0, int l0, int s0, int Lpad,
                                           int h, int mn, int lane, v8f acc) {
  int c = c0 + lane;
  if (c < CH_) {
    const size_t rb = ((size_t)(b * CH_ + c)) * Lpad;
    *(v16h*)&qt[lane * 16] = *(const v16h*)(Qh + rb + l0);
    *(v16h*)&kt[lane * 16] = *(const v16h*)(Kh + rb + s0);
  } else {
    v16h z = {};
    *(v16h*)&qt[lane * 16] = z;
    *(v16h*)&kt[lane * 16] = z;
  }
  asm volatile("s_wait_dscnt 0" ::: "memory");   // intra-wave LDS WAR/RAW fence
  v16h a, bf;
  #pragma unroll
  for (int e = 0; e < 16; ++e) {
    int ka = e + 8 * h + ((e & 8) ? 8 : 0);
    int kb = e + 8 * h + ((e & 8) ? 8 : 0);
    a[e]  = qt[ka * 16 + mn];
    bf[e] = kt[kb * 16 + mn];
  }
  acc = __builtin_amdgcn_wmma_f32_16x16x32_f16(false, a, false, bf, (short)0, acc, false, false);
  asm volatile("" ::: "memory");
  return acc;
}

// ---- scores + reduction: weighted[b,s] = mean_l tanh( (Q^T K)[l,s] / sqrt(L) ) ----
__global__ void scores_weighted_kernel(const _Float16* __restrict__ Qh,
                                       const _Float16* __restrict__ Kh,
                                       int L, int Lpad, float* __restrict__ weighted) {
  __shared__ alignas(32) _Float16 qt[8][512];   // per-wave [c(32)][l(16)]
  __shared__ alignas(32) _Float16 kt[8][512];   // per-wave [c(32)][s(16)]
  __shared__ float red[8][16];
  const int w = threadIdx.x >> 5, lane = threadIdx.x & 31;
  const int h = lane >> 4, mn = lane & 15;
  const int s0 = blockIdx.x * 16;
  const int b  = blockIdx.y;
  const int Ltiles = Lpad >> 4;
  const int nr = (Ltiles + 7) >> 3;
  const float rsl = rsqrtf((float)L);
  float csum = 0.f;
  for (int r = 0; r < nr; ++r) {
    int lt = r * 8 + w;
    if (lt < Ltiles) {                       // wave-uniform predicate: EXEC stays full
      int l0 = lt * 16;
      v8f acc0 = {}, acc1 = {};              // two chains: no WMMA->WMMA hazard nops
      for (int c0 = 0; c0 < CPAD_; c0 += 64) {
        acc0 = scores_step(Qh, Kh, qt[w], kt[w], b, c0,      l0, s0, Lpad, h, mn, lane, acc0);
        acc1 = scores_step(Qh, Kh, qt[w], kt[w], b, c0 + 32, l0, s0, Lpad, h, mn, lane, acc1);
      }
      #pragma unroll
      for (int rr = 0; rr < 8; ++rr) {
        int l = l0 + rr + 8 * h;
        if (l < L) csum += tanhf((acc0[rr] + acc1[rr]) * rsl);
      }
    }
  }
  csum += __shfl_xor(csum, 16, 32);
  if (lane < 16) red[w][lane] = csum;
  __syncthreads();
  if (threadIdx.x < 16) {
    float t = 0.f;
    for (int i = 0; i < 8; ++i) t += red[i][threadIdx.x];
    int s = s0 + threadIdx.x;
    if (s < L) weighted[(size_t)b * Lpad + s] = t / (float)L;
  }
}

// ---- reverse cumsum, softmax(cum*al), win = sum(P*idx) ----
__global__ void attn_win_kernel(const float* __restrict__ weighted, int L, int Lpad,
                                const float* __restrict__ al, float* __restrict__ win) {
  __shared__ float buf[400];
  int b = blockIdx.x;
  if (threadIdx.x == 0) {
    float a = al[0];
    float c = 0.f;
    for (int j = L - 1; j >= 0; --j) { c += weighted[(size_t)b * Lpad + j]; buf[j] = c * a; }
    float mx = -1e30f;
    for (int j = 0; j < L; ++j) mx = fmaxf(mx, buf[j]);
    float se = 0.f;
    for (int j = 0; j < L; ++j) { buf[j] = expf(buf[j] - mx); se += buf[j]; }
    float wv = 0.f;
    for (int j = 0; j < L; ++j) wv += buf[j] * (float)j;
    win[b] = wv / se;
  }
}

__global__ void mask_kernel(const float* __restrict__ win, const float* __restrict__ be,
                            int Lpad, float* __restrict__ maskv) {
  int idx = blockIdx.x * blockDim.x + threadIdx.x;
  if (idx >= B_ * Lpad) return;
  int b = idx / Lpad, j = idx % Lpad;
  maskv[idx] = 1.f / (1.f + expf(-((float)j - win[b]) * be[0]));
}

__global__ void smiw_kernel(const float* __restrict__ iw, int O, float* __restrict__ smiw) {
  int o = threadIdx.x;
  if (o >= O) return;
  float mx = -1e30f;
  for (int c = 0; c < CH_; ++c) mx = fmaxf(mx, iw[(size_t)c * O + o]);
  float s = 0.f;
  for (int c = 0; c < CH_; ++c) s += expf(iw[(size_t)c * O + o] - mx);
  for (int c = 0; c < CH_; ++c) smiw[(size_t)c * O + o] = expf(iw[(size_t)c * O + o] - mx) / s;
}

// ---- pred = smiw * ((comp*mask) @ pw^T + pb + PE) ----
__global__ void pred_kernel(const float* __restrict__ comp, int L,
                            const float* __restrict__ maskv, int Lpad,
                            const float* __restrict__ pw, const float* __restrict__ pb,
                            const float* __restrict__ smiw, int O,
                            float* __restrict__ pred) {
  size_t idx = (size_t)blockIdx.x * blockDim.x + threadIdx.x;
  if (idx >= (size_t)BC_ * O) return;
  int o = (int)(idx % O); size_t r = idx / O;
  int b = (int)(r / CH_), c = (int)(r % CH_);
  float acc = pb[o];
  const float* crow = comp + r * L;
  const float* mrow = maskv + (size_t)b * Lpad;
  const float* wrow = pw + (size_t)o * L;
  for (int j = 0; j < L; ++j) acc += crow[j] * mrow[j] * wrow[j];
  float div = expf(-logf(10000.f) * (float)(o & ~1) / (float)O);
  float pe = (o & 1) ? cosf((float)c * div) : sinf((float)c * div);
  pred[idx] = smiw[(size_t)c * O + o] * (acc + pe);
}

// ---- one iDWT synthesis level (conv_transpose stride 2, pad LF-2) ----
__global__ void sfb_kernel(const float* __restrict__ lo, int lostride,
                           const float* __restrict__ hi, int histride,
                           int Lin, float* __restrict__ out, int ostride) {
  size_t idx = (size_t)blockIdx.x * blockDim.x + threadIdx.x;
  if (idx >= (size_t)BC_ * ostride) return;
  int n = (int)(idx % ostride); size_t bc = idx / ostride;
  const float* lrow = lo + bc * lostride;
  const float* hrow = hi + bc * histride;
  float acc = 0.f;
  #pragma unroll
  for (int t = 0; t < 12; ++t) {
    int j = n + t;
    if (j & 1) {
      int m = (j - 1) >> 1;
      if (m < Lin) acc += lrow[m] * H0c[11 - t] + hrow[m] * H1c[11 - t];
    }
  }
  out[bc * ostride + n] = acc;
}

extern "C" void kernel_launch(void* const* d_in, const int* in_sizes, int n_in,
                              void* d_out, int out_size, void* d_ws, size_t ws_size,
                              hipStream_t stream) {
  (void)in_sizes; (void)n_in; (void)out_size; (void)ws_size;
  const float* x = (const float*)d_in[0];
  const float *QW[4], *QB[4], *KW[4], *KB[4], *PW[4], *PB[4], *IW[4], *AL[4], *BE[4];
  for (int i = 0; i < 4; ++i) {
    const int base = 1 + i * 9;
    QW[i] = (const float*)d_in[base + 0]; QB[i] = (const float*)d_in[base + 1];
    KW[i] = (const float*)d_in[base + 2]; KB[i] = (const float*)d_in[base + 3];
    PW[i] = (const float*)d_in[base + 4]; PB[i] = (const float*)d_in[base + 5];
    IW[i] = (const float*)d_in[base + 6]; AL[i] = (const float*)d_in[base + 7];
    BE[i] = (const float*)d_in[base + 8];
  }
  float* out_ll   = (float*)d_out;
  float* out_mean = out_ll + (size_t)BC_ * LLOUT;
  float* out_std  = out_mean + BC_;

  // bump allocator over workspace
  char* wp_ = (char*)d_ws;
  auto alloc = [&](size_t bytes) { void* r = (void*)wp_; wp_ += (bytes + 255) & ~(size_t)255; return r; };
  float* xc   = (float*)alloc((size_t)BC_ * SEQ_ * 4);
  float* lo1  = (float*)alloc((size_t)BC_ * 365 * 4);
  float* hi1  = (float*)alloc((size_t)BC_ * 365 * 4);
  float* lo2  = (float*)alloc((size_t)BC_ * 188 * 4);
  float* hi2  = (float*)alloc((size_t)BC_ * 188 * 4);
  float* lo3  = (float*)alloc((size_t)BC_ * 99 * 4);
  float* hi3  = (float*)alloc((size_t)BC_ * 99 * 4);
  _Float16* Ah = (_Float16*)alloc((size_t)BC_ * 384 * 2);
  _Float16* Qh = (_Float16*)alloc((size_t)BC_ * 384 * 2);
  _Float16* Kh = (_Float16*)alloc((size_t)BC_ * 384 * 2);
  _Float16* Wp = (_Float16*)alloc((size_t)384 * 384 * 2);
  float* weighted = (float*)alloc((size_t)B_ * 384 * 4);
  float* win   = (float*)alloc((size_t)B_ * 4);
  float* maskv = (float*)alloc((size_t)B_ * 384 * 4);
  float* smiw  = (float*)alloc((size_t)CH_ * 64 * 4);
  const int OL[4] = {22, 57, 34, 22};
  float* preds[4];
  for (int i = 0; i < 4; ++i) preds[i] = (float*)alloc((size_t)BC_ * OL[i] * 4);
  float* t1 = (float*)alloc((size_t)BC_ * 34 * 4);
  float* t2 = (float*)alloc((size_t)BC_ * 58 * 4);

  auto cdiv = [](size_t a, size_t b) { return (unsigned)((a + b - 1) / b); };

  // 1) normalize + transpose
  norm_kernel<<<cdiv(BC_, 256), 256, 0, stream>>>(x, xc, out_mean, out_std);
  // 2) 3-level DWT  (720->365->188->99, left pad 10 each)
  dwt_kernel<<<cdiv((size_t)BC_ * 365, 256), 256, 0, stream>>>(xc, 720, 365, 10, lo1, hi1);
  dwt_kernel<<<cdiv((size_t)BC_ * 188, 256), 256, 0, stream>>>(lo1, 365, 188, 10, lo2, hi2);
  dwt_kernel<<<cdiv((size_t)BC_ * 99, 256), 256, 0, stream>>>(lo2, 188, 99, 10, lo3, hi3);

  const float* comps[4] = {lo3, hi1, hi2, hi3};
  const int    Ls[4]    = {99, 365, 188, 99};
  const int    Lpads[4] = {128, 384, 192, 128};

  for (int i = 0; i < 4; ++i) {
    const int L = Ls[i], Lpad = Lpads[i], O = OL[i];
    const int Ntiles = Lpad >> 4;
    // pack activation to f16
    pack_a_kernel<<<cdiv((size_t)BC_ * Lpad, 256), 256, 0, stream>>>(comps[i], L, Lpad, Ah);
    // Q = tanh(comp @ qw^T + qb)
    pack_w_kernel<<<cdiv((size_t)Lpad * Lpad, 256), 256, 0, stream>>>(QW[i], L, Lpad, Wp);
    gemm_tanh_kernel<<<dim3(BC_ / 128, Ntiles), 256, 0, stream>>>(Ah, Wp, QB[i], L, Lpad, Qh);
    // K = tanh(comp @ kw^T + kb)
    pack_w_kernel<<<cdiv((size_t)Lpad * Lpad, 256), 256, 0, stream>>>(KW[i], L, Lpad, Wp);
    gemm_tanh_kernel<<<dim3(BC_ / 128, Ntiles), 256, 0, stream>>>(Ah, Wp, KB[i], L, Lpad, Kh);
    // weighted[b,s] = mean_l tanh(scores/sqrt(L))
    scores_weighted_kernel<<<dim3(Ntiles, B_), 256, 0, stream>>>(Qh, Kh, L, Lpad, weighted);
    // reverse cumsum / softmax / window
    attn_win_kernel<<<B_, 32, 0, stream>>>(weighted, L, Lpad, AL[i], win);
    mask_kernel<<<cdiv((size_t)B_ * Lpad, 256), 256, 0, stream>>>(win, BE[i], Lpad, maskv);
    // softmax(iw) over channels, then projection + PE
    smiw_kernel<<<1, 64, 0, stream>>>(IW[i], O, smiw);
    pred_kernel<<<cdiv((size_t)BC_ * O, 256), 256, 0, stream>>>(comps[i], L, maskv, Lpad,
                                                                PW[i], PB[i], smiw, O, preds[i]);
  }

  // 3) inverse DWT: 22 -> 34 -> 58(trunc 57) -> 104, last stage straight into d_out
  sfb_kernel<<<cdiv((size_t)BC_ * 34, 256), 256, 0, stream>>>(preds[0], 22, preds[3], 22, 22, t1, 34);
  sfb_kernel<<<cdiv((size_t)BC_ * 58, 256), 256, 0, stream>>>(t1, 34, preds[2], 34, 34, t2, 58);
  sfb_kernel<<<cdiv((size_t)BC_ * LLOUT, 256), 256, 0, stream>>>(t2, 58, preds[1], 57, 57, out_ll, LLOUT);
}
